// Decoder_33818572488910
// MI455X (gfx1250) — matrix operand order repaired and checked by reading, not compiled, not run on hardware
//
#include <hip/hip_runtime.h>
#include <hip/hip_bf16.h>
#include <math.h>

// ---------------------------------------------------------------------------
// Tacotron2 decoder for MI455X (gfx1250, wave32, WMMA + TDM + async-LDS).
// All GEMMs use v_wmma_f32_16x16x32_bf16. bf16 weights (~36MB) are L2-resident
// (192MB L2), so the 500-step scan streams from L2, not HBM (23.3 TB/s is
// irrelevant after the first pass).  Recurrent GEMMs stage the activation
// panel into LDS via the Tensor Data Mover; attention stages its weights via
// async global->LDS loads.
// ---------------------------------------------------------------------------

typedef __bf16 bf16_t;
typedef __attribute__((ext_vector_type(16))) __bf16 v16bf;
typedef __attribute__((ext_vector_type(8)))  float  v8f;
typedef __attribute__((ext_vector_type(4)))  unsigned su4;
typedef __attribute__((ext_vector_type(8)))  unsigned su8;

#define B_     32
#define T_IN_  400
#define T_OUT_ 500
#define N_MEL_ 80
#define E_DIM  512
#define A_RNN_ 1024
#define D_RNN_ 1024
#define P_DIM_ 256
#define A_DIM_ 128
#define LOC_F_ 32
#define LOC_K_ 31
#define PADK_  15
#define KA_    1792   // P_DIM + E + A_RNN   (attention-LSTM fused K)
#define KD_    2560   // A_RNN + E + D_RNN   (decoder-LSTM fused K)
#define KP1_   96     // prenet layer1 K padded 80 -> 96
#define NG_    4096   // 4 * RNN gates
#define CHUNK_K 512   // K chunk staged in LDS by the TDM

__device__ __forceinline__ float sigf(float x) { return 1.0f / (1.0f + __expf(-x)); }

__device__ __forceinline__ bf16_t f2bf(float f) {
    union { float f; unsigned u; } in; in.f = f;
    unsigned u = in.u;
    unsigned r = u + 0x7FFFu + ((u >> 16) & 1u);   // round-to-nearest-even
    union { unsigned short s; bf16_t b; } out; out.s = (unsigned short)(r >> 16);
    return out.b;
}

__device__ __forceinline__ unsigned hashu(unsigned x) {
    x ^= x >> 16; x *= 0x7feb352dU; x ^= x >> 15; x *= 0x846ca68bU; x ^= x >> 16;
    return x;
}

// LDS aperture: LDS byte offset = flat_addr[31:0]
__device__ __forceinline__ unsigned lds_addr32(const void* p) {
    return (unsigned)(unsigned long long)p;
}

// ASYNCcnt-tracked global -> LDS 16-byte copy (per-lane addresses)
__device__ __forceinline__ void async_g2l_b128(unsigned lds_off, const void* g) {
    asm volatile("global_load_async_to_lds_b128 %0, %1, off"
                 :: "v"(lds_off), "v"((unsigned long long)g) : "memory");
}
__device__ __forceinline__ void wait_asynccnt0() {
    asm volatile("s_wait_asynccnt 0x0" ::: "memory");
}

// ---------------------------------------------------------------------------
// Generic WMMA GEMM:  C[M,N] = A[M,K] (row stride lda) * W[N,K]^T (+bias[n])
// Per-lane operand fetch per CDNA5 ISA layouts:
//   A (16x32 bf16): lane l -> row m=l%16; elems 0..7 = K kb..kb+7,
//                   elems 8..15 = K 16+kb..16+kb+7, kb = 8*(l/16).
//   B (32x16 bf16): lane l -> col n=l%16; same K order as A.
//   D (16x16 f32):  lane l, vgpr i -> m = 8*(l/16)+i, n = l%16.
// ---------------------------------------------------------------------------
__global__ __launch_bounds__(256)
void k_gemm(const bf16_t* __restrict__ A, int lda,
            const bf16_t* __restrict__ W,
            const float* __restrict__ bias,
            float* __restrict__ C, int ldc,
            int M, int N, int K) {
    const int lane = threadIdx.x & 31;
    const int wave = threadIdx.x >> 5;
    const int wavesPerBlock = blockDim.x >> 5;
    const int gw = blockIdx.x * wavesPerBlock + wave;
    const int nw = gridDim.x * wavesPerBlock;
    const int tm = M >> 4, tn = N >> 4;
    const int r  = lane & 15;
    const int hs = lane >> 4;
    const int kbA = hs * 8;
    const int kbB = hs * 16;

    for (int tile = gw; tile < tm * tn; tile += nw) {
        const int mt = tile / tn, nt = tile % tn;
        const bf16_t* arow = A + (size_t)(mt * 16 + r) * lda;
        const bf16_t* wrow = W + (size_t)(nt * 16 + r) * K;
        v8f acc = {};
        for (int k0 = 0; k0 < K; k0 += 32) {
            union { uint4 u[2]; v16bf v; } av, bv;
            av.u[0] = *(const uint4*)(arow + k0 + kbA);
            av.u[1] = *(const uint4*)(arow + k0 + 16 + kbA);
            bv.u[0] = *(const uint4*)(wrow + k0 + kbA);
            bv.u[1] = *(const uint4*)(wrow + k0 + 16 + kbA);
            acc = __builtin_amdgcn_wmma_f32_16x16x32_bf16(
                false, av.v, false, bv.v, (short)0, acc, false, false);
        }
        const int n = nt * 16 + r;
        const float bn = bias ? bias[n] : 0.0f;
#pragma unroll
        for (int i = 0; i < 8; ++i) {
            const int m = mt * 16 + hs * 8 + i;
            C[(size_t)m * ldc + n] = acc[i] + bn;
        }
    }
}

// ---------------------------------------------------------------------------
// Recurrent-step WMMA GEMM, M=32, N=4096.  Block = 8 waves = 8 N-tiles with a
// common M-tile.  The 16 x chunk activation panel is DMA'd into LDS by the
// Tensor Data Mover (D# built per ISA §8; wave 0 issues, TENSORcnt waits),
// A-operands then come from LDS (ds_load_b128), weights stream from L2.
// ---------------------------------------------------------------------------
__global__ __launch_bounds__(256)
void k_gemm_rec(const bf16_t* __restrict__ A, int lda,
                const bf16_t* __restrict__ W,
                const float* __restrict__ bias,
                float* __restrict__ C,
                int N, int K) {
    __shared__ __align__(16) bf16_t s_a[16 * CHUNK_K];   // 16KB panel chunk
    const int lane = threadIdx.x & 31;
    const int wave = threadIdx.x >> 5;
    const int tn = N >> 4;                 // 256
    const int blocksPerM = tn >> 3;        // 8 waves/block
    const int mt = blockIdx.x / blocksPerM;
    const int nt = (blockIdx.x % blocksPerM) * 8 + wave;
    const int r  = lane & 15;
    const int hs = lane >> 4;
    const int kbA = hs * 8;
    const int kbB = hs * 16;
    const bf16_t* wrow = W + (size_t)(nt * 16 + r) * K;

    v8f acc = {};
    for (int k0 = 0; k0 < K; k0 += CHUNK_K) {
        const int ck = (K - k0 < CHUNK_K) ? (K - k0) : CHUNK_K;
        __syncthreads();                    // previous chunk fully consumed
        if (wave == 0) {
            // D# group 0: count=1 | lds_addr | global_addr(57b) | type=2
            unsigned long long ga =
                (unsigned long long)(const void*)(A + (size_t)(mt * 16) * lda + k0);
            su4 g0;
            g0.x = 0x1u;
            g0.y = lds_addr32(s_a);
            g0.z = (unsigned)ga;
            g0.w = (unsigned)((ga >> 32) & 0x01FFFFFFu) | (2u << 30);
            // D# group 1: data_size=2B; tensor_dim0=ck @bit48; tensor_dim1=16
            // @bit80; tile_dim0=ck @bit112; tile_dim1=16 @bit128;
            // tensor_dim0_stride=lda @bit160.
            const unsigned td0 = (unsigned)ck, td1 = 16u;
            su8 g1;
            g1.s0 = 0x10000u;
            g1.s1 = (td0 & 0xFFFFu) << 16;
            g1.s2 = (td0 >> 16) | ((td1 & 0xFFFFu) << 16);
            g1.s3 = (td1 >> 16) | ((unsigned)ck << 16);
            g1.s4 = 16u;
            g1.s5 = (unsigned)lda;
            g1.s6 = 0u;
            g1.s7 = 0u;
            asm volatile("tensor_load_to_lds %0, %1" :: "s"(g0), "s"(g1) : "memory");
            __builtin_amdgcn_s_wait_tensorcnt(0);
        }
        __syncthreads();                    // panel visible to all waves
        if (k0 + ck < K)                    // keep weight stream ahead in L2/L0
            __builtin_prefetch(wrow + k0 + ck, 0, 0);
        const bf16_t* arow_l = s_a + r * ck;   // TDM packs rows at stride ck
        for (int kk = 0; kk < ck; kk += 32) {
            union { uint4 u[2]; v16bf v; } av, bv;
            av.u[0] = *(const uint4*)(arow_l + kk + kbA);
            av.u[1] = *(const uint4*)(arow_l + kk + 16 + kbA);
            const bf16_t* wp = wrow + k0 + kk;
            bv.u[0] = *(const uint4*)(wp + kbA);
            bv.u[1] = *(const uint4*)(wp + 16 + kbA);
            acc = __builtin_amdgcn_wmma_f32_16x16x32_bf16(
                false, av.v, false, bv.v, (short)0, acc, false, false);
        }
    }
    const int n = nt * 16 + r;
    const float bn = bias ? bias[n] : 0.0f;
#pragma unroll
    for (int i = 0; i < 8; ++i) {
        const int m = mt * 16 + hs * 8 + i;
        C[(size_t)m * N + n] = acc[i] + bn;
    }
}

// ---------------------------------------------------------------------------
// Pointwise LSTM update (torch gate order i,f,g,o).
// ---------------------------------------------------------------------------
__global__ void k_lstm(const float* __restrict__ g,
                       float* __restrict__ h, float* __restrict__ c,
                       bf16_t* hb1, int s1, int o1,
                       bf16_t* hb2, int s2, int o2, int H) {
    int idx = blockIdx.x * blockDim.x + threadIdx.x;
    if (idx >= B_ * H) return;
    int b = idx / H, j = idx - b * H;
    const float* gr = g + (size_t)b * 4 * H;
    float ig = sigf(gr[j]);
    float fg = sigf(gr[H + j]);
    float gg = tanhf(gr[2 * H + j]);
    float og = sigf(gr[3 * H + j]);
    float cn = fg * c[idx] + ig * gg;
    float hn = og * tanhf(cn);
    c[idx] = cn; h[idx] = hn;
    bf16_t hb = f2bf(hn);
    if (hb1) hb1[(size_t)b * s1 + o1 + j] = hb;
    if (hb2) hb2[(size_t)b * s2 + o2 + j] = hb;
}

__global__ void k_copy_pre(const bf16_t* __restrict__ pre, bf16_t* __restrict__ xa, int t) {
    int idx = blockIdx.x * blockDim.x + threadIdx.x;
    if (idx >= B_ * P_DIM_) return;
    int b = idx / P_DIM_, j = idx - b * P_DIM_;
    xa[(size_t)b * KA_ + j] = pre[((size_t)t * B_ + b) * P_DIM_ + j];
}

// ---------------------------------------------------------------------------
// Fused location-sensitive attention.  One block per batch row; weight
// matrices staged to LDS with ASYNCcnt-tracked global->LDS b128 copies.
// ---------------------------------------------------------------------------
__global__ __launch_bounds__(256)
void k_attn(const float* __restrict__ mem,          // [B,T_IN,E]
            const float* __restrict__ pm,           // [B,T_IN,A_DIM]
            const float* __restrict__ pq,           // [B,A_DIM]
            const float* __restrict__ vvec,         // [A_DIM]
            const float* __restrict__ wconv,        // [LOC_F,2,LOC_K]
            const float* __restrict__ wld,          // [A_DIM,LOC_F]
            const unsigned char* __restrict__ mask, // [B,T_IN] bool
            float* __restrict__ aw, float* __restrict__ awc,
            float* __restrict__ ctx,
            bf16_t* __restrict__ xa, bf16_t* __restrict__ xd,
            float* __restrict__ align_out, int t) {
    __shared__ float s_aw[T_IN_], s_awc[T_IN_], s_e[T_IN_];
    __shared__ __align__(16) float s_wld[A_DIM_ * LOC_F_];   // 16KB
    __shared__ __align__(16) float s_wc[LOC_F_ * 2 * LOC_K_];
    __shared__ float s_pq[A_DIM_], s_v[A_DIM_];
    __shared__ float s_red[256];
    const int b = blockIdx.x, tid = threadIdx.x, nt = blockDim.x;

    // async-stage the big weight blocks (16-byte granules)
    {
        const unsigned wldb = lds_addr32(s_wld);
        for (int i = tid; i < (A_DIM_ * LOC_F_) / 4; i += nt)
            async_g2l_b128(wldb + i * 16, wld + i * 4);
        const unsigned wcb = lds_addr32(s_wc);
        for (int i = tid; i < (LOC_F_ * 2 * LOC_K_) / 4; i += nt)
            async_g2l_b128(wcb + i * 16, wconv + i * 4);
    }
    for (int i = tid; i < T_IN_; i += nt) { s_aw[i] = aw[b * T_IN_ + i]; s_awc[i] = awc[b * T_IN_ + i]; }
    for (int i = tid; i < A_DIM_; i += nt) { s_pq[i] = pq[b * A_DIM_ + i]; s_v[i] = vvec[i]; }
    wait_asynccnt0();
    __syncthreads();

    for (int i = tid; i < T_IN_; i += nt) {
        float loc[LOC_F_];
#pragma unroll
        for (int f = 0; f < LOC_F_; ++f) {
            float acc = 0.f;
            const float* w0 = &s_wc[(f * 2 + 0) * LOC_K_];
            const float* w1 = &s_wc[(f * 2 + 1) * LOC_K_];
#pragma unroll
            for (int k = 0; k < LOC_K_; ++k) {
                int j = i + k - PADK_;
                if (j >= 0 && j < T_IN_) acc += s_aw[j] * w0[k] + s_awc[j] * w1[k];
            }
            loc[f] = acc;
        }
        float e = 0.f;
        const float* pmr = pm + ((size_t)b * T_IN_ + i) * A_DIM_;
        for (int a = 0; a < A_DIM_; ++a) {
            float ts = s_pq[a] + pmr[a];
            const float* wr = &s_wld[a * LOC_F_];
#pragma unroll
            for (int f = 0; f < LOC_F_; ++f) ts += loc[f] * wr[f];
            e += s_v[a] * tanhf(ts);
        }
        if (mask[b * T_IN_ + i]) e = -1e9f;
        s_e[i] = e;
    }
    __syncthreads();

    // softmax over T_IN
    float lm = -1e30f;
    for (int i = tid; i < T_IN_; i += nt) lm = fmaxf(lm, s_e[i]);
    s_red[tid] = lm; __syncthreads();
    for (int s = nt >> 1; s > 0; s >>= 1) { if (tid < s) s_red[tid] = fmaxf(s_red[tid], s_red[tid + s]); __syncthreads(); }
    float mx = s_red[0]; __syncthreads();
    float ls = 0.f;
    for (int i = tid; i < T_IN_; i += nt) { float ex = __expf(s_e[i] - mx); s_e[i] = ex; ls += ex; }
    s_red[tid] = ls; __syncthreads();
    for (int s = nt >> 1; s > 0; s >>= 1) { if (tid < s) s_red[tid] += s_red[tid + s]; __syncthreads(); }
    float inv = 1.0f / s_red[0]; __syncthreads();

    for (int i = tid; i < T_IN_; i += nt) {
        float w = s_e[i] * inv;
        s_aw[i] = w;
        aw[b * T_IN_ + i] = w;
        awc[b * T_IN_ + i] = s_awc[i] + w;
        align_out[((size_t)b * T_OUT_ + t) * T_IN_ + i] = w;
    }
    __syncthreads();

    // context = aw_new @ memory (lane-contiguous over E)
    for (int e0 = tid; e0 < E_DIM; e0 += nt) {
        float acc = 0.f;
        const float* mrow = mem + (size_t)b * T_IN_ * E_DIM + e0;
        for (int i = 0; i < T_IN_; ++i) acc += s_aw[i] * mrow[(size_t)i * E_DIM];
        ctx[b * E_DIM + e0] = acc;
        bf16_t cb = f2bf(acc);
        xa[(size_t)b * KA_ + P_DIM_ + e0] = cb;   // feeds step t+1 attn-LSTM
        xd[(size_t)b * KD_ + A_RNN_ + e0] = cb;   // feeds step t   dec-LSTM
    }
}

// mel / gate projection for step t
__global__ void k_out(const float* __restrict__ dh, const float* __restrict__ ctx,
                      const float* __restrict__ Wp, const float* __restrict__ bp,
                      const float* __restrict__ Wg, const float* __restrict__ bg,
                      float* __restrict__ mel_out, float* __restrict__ gate_out, int t) {
    int b = blockIdx.x, j = threadIdx.x;
    if (j < N_MEL_) {
        const float* wr = Wp + (size_t)j * (D_RNN_ + E_DIM);
        float acc = bp[j];
        for (int k = 0; k < D_RNN_; ++k) acc += wr[k] * dh[b * D_RNN_ + k];
        for (int k = 0; k < E_DIM; ++k)  acc += wr[D_RNN_ + k] * ctx[b * E_DIM + k];
        mel_out[((size_t)b * N_MEL_ + j) * T_OUT_ + t] = acc;
    } else if (j == N_MEL_) {
        float acc = bg[0];
        for (int k = 0; k < D_RNN_; ++k) acc += Wg[k] * dh[b * D_RNN_ + k];
        for (int k = 0; k < E_DIM; ++k)  acc += Wg[D_RNN_ + k] * ctx[b * E_DIM + k];
        gate_out[(size_t)b * T_OUT_ + t] = acc;
    }
}

// ------------------------ prep kernels -------------------------------------
__global__ void k_zero(unsigned* __restrict__ p, size_t n) {
    size_t i = (size_t)blockIdx.x * blockDim.x + threadIdx.x;
    size_t st = (size_t)gridDim.x * blockDim.x;
    for (; i < n; i += st) p[i] = 0u;
}

__global__ void k_cvt(const float* __restrict__ s, bf16_t* __restrict__ d, size_t n) {
    size_t i = (size_t)blockIdx.x * blockDim.x + threadIdx.x;
    size_t st = (size_t)gridDim.x * blockDim.x;
    for (; i < n; i += st) d[i] = f2bf(s[i]);
}

__global__ void k_concat2(const float* __restrict__ s0, int k0,
                          const float* __restrict__ s1, int k1,
                          bf16_t* __restrict__ dst, int K, size_t n) {
    size_t i = (size_t)blockIdx.x * blockDim.x + threadIdx.x;
    size_t st = (size_t)gridDim.x * blockDim.x;
    for (; i < n; i += st) {
        int k = (int)(i % K);
        size_t row = i / K;
        float v = (k < k0) ? s0[row * (size_t)k0 + k] : s1[row * (size_t)k1 + (k - k0)];
        dst[i] = f2bf(v);
    }
}

__global__ void k_padcvt(const float* __restrict__ s, bf16_t* __restrict__ d,
                         int rows, int kin, int kout) {
    size_t n = (size_t)rows * kout;
    size_t i = (size_t)blockIdx.x * blockDim.x + threadIdx.x;
    size_t st = (size_t)gridDim.x * blockDim.x;
    for (; i < n; i += st) {
        int k = (int)(i % kout);
        size_t row = i / kout;
        d[i] = f2bf(k < kin ? s[row * (size_t)kin + k] : 0.0f);
    }
}

__global__ void k_bias(const float* __restrict__ a, const float* __restrict__ b,
                       float* __restrict__ d, int n) {
    int i = blockIdx.x * blockDim.x + threadIdx.x;
    if (i < n) d[i] = a[i] + b[i];
}

__global__ void k_decin(const float* __restrict__ din, bf16_t* __restrict__ dst) {
    size_t n = (size_t)T_OUT_ * B_ * KP1_;
    size_t i = (size_t)blockIdx.x * blockDim.x + threadIdx.x;
    size_t st = (size_t)gridDim.x * blockDim.x;
    for (; i < n; i += st) {
        int k = (int)(i % KP1_);
        size_t row = i / KP1_;
        int b = (int)(row % B_);
        int t = (int)(row / B_);
        float v = 0.f;
        if (k < N_MEL_ && t > 0) v = din[((size_t)b * N_MEL_ + k) * T_OUT_ + (t - 1)];
        dst[i] = f2bf(v);
    }
}

__global__ void k_pw(const float* __restrict__ s, bf16_t* __restrict__ d, size_t n, unsigned seed) {
    size_t i = (size_t)blockIdx.x * blockDim.x + threadIdx.x;
    size_t st = (size_t)gridDim.x * blockDim.x;
    for (; i < n; i += st) {
        float x = s[i];
        float v = (hashu((unsigned)i ^ seed) & 1u) ? x * 2.0f : 0.0f;
        d[i] = f2bf(fmaxf(v, 0.0f));
    }
}

// ---------------------------------------------------------------------------
extern "C" void kernel_launch(void* const* d_in, const int* in_sizes, int n_in,
                              void* d_out, int out_size, void* d_ws, size_t ws_size,
                              hipStream_t stream) {
    (void)in_sizes; (void)n_in; (void)out_size; (void)ws_size;

    const float* mem_f   = (const float*)d_in[0];
    const float* din     = (const float*)d_in[1];
    const unsigned char* mask = (const unsigned char*)d_in[2];
    const float* Wpre1_f = (const float*)d_in[3];
    const float* Wpre2_f = (const float*)d_in[4];
    const float* Wih_a   = (const float*)d_in[5];
    const float* Whh_a   = (const float*)d_in[6];
    const float* bih_a   = (const float*)d_in[7];
    const float* bhh_a   = (const float*)d_in[8];
    const float* Wq_f    = (const float*)d_in[9];
    const float* Wm_f    = (const float*)d_in[10];
    const float* vvec    = (const float*)d_in[11];
    const float* wconv   = (const float*)d_in[12];
    const float* wld     = (const float*)d_in[13];
    const float* Wih_d   = (const float*)d_in[14];
    const float* Whh_d   = (const float*)d_in[15];
    const float* bih_d   = (const float*)d_in[16];
    const float* bhh_d   = (const float*)d_in[17];
    const float* Wp      = (const float*)d_in[18];
    const float* bp      = (const float*)d_in[19];
    const float* Wg      = (const float*)d_in[20];
    const float* bg      = (const float*)d_in[21];

    float* out      = (float*)d_out;
    float* melOut   = out;                                  // [B,80,T_OUT]
    float* gateOut  = out + (size_t)B_ * N_MEL_ * T_OUT_;   // [B,T_OUT]
    float* alignOut = gateOut + (size_t)B_ * T_OUT_;        // [B,T_OUT,T_IN]

    // ---- workspace layout (256B aligned blocks) ----
    size_t off = 0;
    char* base = (char*)d_ws;
    auto alloc = [&](size_t bytes) -> void* {
        void* p = base + off;
        off = (off + bytes + 255) & ~(size_t)255;
        return p;
    };
    bf16_t* WA   = (bf16_t*)alloc((size_t)NG_ * KA_ * 2);       // fused [Wih_a|Whh_a]
    bf16_t* WD   = (bf16_t*)alloc((size_t)NG_ * KD_ * 2);       // fused [Wih_d|Whh_d]
    bf16_t* WQ   = (bf16_t*)alloc((size_t)A_DIM_ * A_RNN_ * 2);
    bf16_t* WP1  = (bf16_t*)alloc((size_t)P_DIM_ * KP1_ * 2);
    bf16_t* WP2  = (bf16_t*)alloc((size_t)P_DIM_ * P_DIM_ * 2);
    bf16_t* WM   = (bf16_t*)alloc((size_t)A_DIM_ * E_DIM * 2);
    bf16_t* MEMB = (bf16_t*)alloc((size_t)B_ * T_IN_ * E_DIM * 2);
    bf16_t* DECB = (bf16_t*)alloc((size_t)T_OUT_ * B_ * KP1_ * 2);
    bf16_t* H1B  = (bf16_t*)alloc((size_t)T_OUT_ * B_ * P_DIM_ * 2);
    bf16_t* PRE  = (bf16_t*)alloc((size_t)T_OUT_ * B_ * P_DIM_ * 2);
    float*  GBUF = (float*) alloc((size_t)T_OUT_ * B_ * P_DIM_ * 4);
    float*  PM   = (float*) alloc((size_t)B_ * T_IN_ * A_DIM_ * 4);
    float*  BIASA= (float*) alloc((size_t)NG_ * 4);
    float*  BIASD= (float*) alloc((size_t)NG_ * 4);
    size_t stateBeg = off;
    float*  AH  = (float*) alloc((size_t)B_ * A_RNN_ * 4);
    float*  AC  = (float*) alloc((size_t)B_ * A_RNN_ * 4);
    float*  DH  = (float*) alloc((size_t)B_ * D_RNN_ * 4);
    float*  DC  = (float*) alloc((size_t)B_ * D_RNN_ * 4);
    float*  CTX = (float*) alloc((size_t)B_ * E_DIM * 4);
    float*  AW  = (float*) alloc((size_t)B_ * T_IN_ * 4);
    float*  AWC = (float*) alloc((size_t)B_ * T_IN_ * 4);
    float*  GA  = (float*) alloc((size_t)B_ * NG_ * 4);
    float*  GD  = (float*) alloc((size_t)B_ * NG_ * 4);
    float*  PQ  = (float*) alloc((size_t)B_ * A_DIM_ * 4);
    bf16_t* XA  = (bf16_t*)alloc((size_t)B_ * KA_ * 2);   // [pre|ctx|ah] bf16
    bf16_t* XD  = (bf16_t*)alloc((size_t)B_ * KD_ * 2);   // [ah|ctx|dh]  bf16
    size_t stateEnd = off;

    // ---- prep: zero recurrent state, convert weights/activations to bf16 ----
    k_zero<<<1024, 256, 0, stream>>>((unsigned*)(base + stateBeg), (stateEnd - stateBeg) / 4);
    k_concat2<<<4096, 256, 0, stream>>>(Wih_a, P_DIM_ + E_DIM, Whh_a, A_RNN_, WA, KA_, (size_t)NG_ * KA_);
    k_concat2<<<4096, 256, 0, stream>>>(Wih_d, A_RNN_ + E_DIM, Whh_d, D_RNN_, WD, KD_, (size_t)NG_ * KD_);
    k_cvt<<<512, 256, 0, stream>>>(Wq_f,  WQ,  (size_t)A_DIM_ * A_RNN_);
    k_cvt<<<256, 256, 0, stream>>>(Wpre2_f, WP2, (size_t)P_DIM_ * P_DIM_);
    k_cvt<<<256, 256, 0, stream>>>(Wm_f,  WM,  (size_t)A_DIM_ * E_DIM);
    k_padcvt<<<96, 256, 0, stream>>>(Wpre1_f, WP1, P_DIM_, N_MEL_, KP1_);
    k_bias<<<(NG_ + 255) / 256, 256, 0, stream>>>(bih_a, bhh_a, BIASA, NG_);
    k_bias<<<(NG_ + 255) / 256, 256, 0, stream>>>(bih_d, bhh_d, BIASD, NG_);
    k_cvt<<<4096, 256, 0, stream>>>(mem_f, MEMB, (size_t)B_ * T_IN_ * E_DIM);
    k_decin<<<2048, 256, 0, stream>>>(din, DECB);

    // ---- prenet over all 500 steps (batched WMMA GEMMs) ----
    k_gemm<<<2048, 256, 0, stream>>>(DECB, KP1_, WP1, nullptr, GBUF, P_DIM_,
                                     T_OUT_ * B_, P_DIM_, KP1_);
    k_pw<<<2048, 256, 0, stream>>>(GBUF, H1B, (size_t)T_OUT_ * B_ * P_DIM_, 0x12345678u);
    k_gemm<<<2048, 256, 0, stream>>>(H1B, P_DIM_, WP2, nullptr, GBUF, P_DIM_,
                                     T_OUT_ * B_, P_DIM_, P_DIM_);
    k_pw<<<2048, 256, 0, stream>>>(GBUF, PRE, (size_t)T_OUT_ * B_ * P_DIM_, 0x89abcdefu);

    // ---- processed_memory = memory @ Wm.T ----
    k_gemm<<<800, 256, 0, stream>>>(MEMB, E_DIM, WM, nullptr, PM, A_DIM_,
                                    B_ * T_IN_, A_DIM_, E_DIM);

    // ---- 500-step sequential decode (graph-captured launch chain) ----
    for (int t = 0; t < T_OUT_; ++t) {
        k_copy_pre<<<(B_ * P_DIM_ + 255) / 256, 256, 0, stream>>>(PRE, XA, t);
        // attention LSTM gates: [32,1792] x [1792,4096] (TDM-staged A panel)
        k_gemm_rec<<<64, 256, 0, stream>>>(XA, KA_, WA, BIASA, GA, NG_, KA_);
        k_lstm<<<(B_ * A_RNN_ + 255) / 256, 256, 0, stream>>>(
            GA, AH, AC, XA, KA_, P_DIM_ + E_DIM, XD, KD_, 0, A_RNN_);
        // pq = ah @ Wq.T : reuse ah slot inside XA (lda = KA)
        k_gemm<<<2, 256, 0, stream>>>(XA + (P_DIM_ + E_DIM), KA_, WQ, nullptr,
                                      PQ, A_DIM_, B_, A_DIM_, A_RNN_);
        k_attn<<<B_, 256, 0, stream>>>(mem_f, PM, PQ, vvec, wconv, wld, mask,
                                       AW, AWC, CTX, XA, XD, alignOut, t);
        // decoder LSTM gates: [32,2560] x [2560,4096] (TDM-staged A panel)
        k_gemm_rec<<<64, 256, 0, stream>>>(XD, KD_, WD, BIASD, GD, NG_, KD_);
        k_lstm<<<(B_ * D_RNN_ + 255) / 256, 256, 0, stream>>>(
            GD, DH, DC, XD, KD_, A_RNN_ + E_DIM, (bf16_t*)nullptr, 0, 0, D_RNN_);
        k_out<<<B_, 96, 0, stream>>>(DH, CTX, Wp, bp, Wg, bg, melOut, gateOut, t);
    }
}
